// ScaledDotProductAttention_TA_1786706395421
// MI455X (gfx1250) — matrix operand order repaired and checked by reading, not compiled, not run on hardware
//
#include <hip/hip_runtime.h>
#include <math.h>

#define B_ 16
#define L_ 2048
#define D_ 128

typedef __attribute__((ext_vector_type(16))) __bf16        bf16x16;
typedef __attribute__((ext_vector_type(2)))  __bf16        bf16x2;
typedef __attribute__((ext_vector_type(16))) float         f32x16;
typedef __attribute__((ext_vector_type(8)))  float         f32x8;
typedef __attribute__((ext_vector_type(4)))  float         f32x4;
typedef __attribute__((ext_vector_type(2)))  float         f32x2;
typedef __attribute__((ext_vector_type(4)))  unsigned int  u32x4;

union Frag {
  bf16x16      bf;
  unsigned int u[8];
  u32x4        q4[2];
};

// Paired fp32 -> packed-bf16 conversion (lowers to v_cvt_pk_bf16_f32 a,b).
__device__ __forceinline__ unsigned int pk2(float a, float b) {
  f32x2 t; t[0] = a; t[1] = b;
  union { bf16x2 h; unsigned int u; } x;
  x.h = __builtin_convertvector(t, bf16x2);
  return x.u;
}
__device__ __forceinline__ unsigned short bfbits(float a) {
  union { __bf16 h; unsigned short s; } x;
  x.h = (__bf16)a; return x.s;
}

__device__ __forceinline__ f32x8 wmma_bf16(const Frag& a, const Frag& b, f32x8 c) {
  return __builtin_amdgcn_wmma_f32_16x16x32_bf16(false, a.bf, false, b.bf,
                                                 (short)0, c, false, false);
}

// A fragment (16x32 bf16) built from an fp32 row (this lane's row), kstep t.
// Per-lane K pattern: {base..base+7, base+16..base+23}, base = 32t + 8*half.
// All 16 values are converted with ONE vector convert so the backend emits
// 8 paired v_cvt_pk_bf16_f32 straight into the fragment VGPRs.
__device__ __forceinline__ Frag a_frag_f32(const float* __restrict__ rowp, int t,
                                           int half, float scale) {
  const int base = t * 32 + half * 8;
  const f32x4 x0 = *(const f32x4*)(rowp + base);
  const f32x4 x1 = *(const f32x4*)(rowp + base + 4);
  const f32x4 x2 = *(const f32x4*)(rowp + base + 16);
  const f32x4 x3 = *(const f32x4*)(rowp + base + 20);
  f32x16 w;
  w[0]  = x0[0]; w[1]  = x0[1]; w[2]  = x0[2]; w[3]  = x0[3];
  w[4]  = x1[0]; w[5]  = x1[1]; w[6]  = x1[2]; w[7]  = x1[3];
  w[8]  = x2[0]; w[9]  = x2[1]; w[10] = x2[2]; w[11] = x2[3];
  w[12] = x3[0]; w[13] = x3[1]; w[14] = x3[2]; w[15] = x3[3];
  w = w * scale;
  Frag f;
  f.bf = __builtin_convertvector(w, bf16x16);
  return f;
}

// A fragment from a bf16 row (global ws or LDS): two b128 loads.
__device__ __forceinline__ Frag a_frag_bf(const unsigned short* rowp, int t, int half) {
  Frag f;
  const int base = t * 32 + half * 8;
  f.q4[0] = *(const u32x4*)(rowp + base);
  f.q4[1] = *(const u32x4*)(rowp + base + 16);
  return f;
}

// B fragment (32x16): per lane K values {base..base+7, base+16..base+23}, base = 32t + 8*khalf,
// colp points at this lane's column data with K contiguous.
__device__ __forceinline__ Frag b_frag_bf(const unsigned short* colp, int t, int khalf) {
  Frag f;
  const int base = t * 32 + khalf * 8;
  f.q4[0] = *(const u32x4*)(colp + base);
  f.q4[1] = *(const u32x4*)(colp + base + 16);
  return f;
}

// ---------------------------------------------------------------------------
// Kernel 1: q2w = (dw @ q2)/T, k2w = dw @ k2  (bf16 outputs to workspace)
// grid = (L/128, B, 2), block = 256 (8 waves). Each wave owns 16 rows x 128 cols.
// ---------------------------------------------------------------------------
__global__ __launch_bounds__(256)
void proj_kernel(const float* __restrict__ dw, const float* __restrict__ q2,
                 const float* __restrict__ k2,
                 unsigned short* __restrict__ q2w, unsigned short* __restrict__ k2w) {
  __shared__ __align__(16) unsigned short Xt[D_][72];  // 64-row X chunk, transposed

  const int b    = blockIdx.y;
  const int sel  = blockIdx.z;
  const int m0   = blockIdx.x * 128;
  const int tid  = threadIdx.x;
  const int wave = tid >> 5, lane = tid & 31;
  const int half = lane >> 4, lcol = lane & 15;

  const float* X   = (sel ? k2 : q2) + (size_t)b * L_ * D_;
  const float* dwr = dw + (size_t)b * L_ * L_ + (size_t)(m0 + wave * 16 + lcol) * L_;

  f32x8 acc[8];
#pragma unroll
  for (int c = 0; c < 8; ++c)
#pragma unroll
    for (int r = 0; r < 8; ++r) acc[c][r] = 0.0f;

  for (int kc = 0; kc < L_ / 64; ++kc) {
    __syncthreads();
    // Stage 64 rows of X transposed; process dim-pairs with paired converts.
#pragma unroll 4
    for (int i = tid; i < 64 * 64; i += 256) {
      const int ll = i >> 6, dp = (i & 63) * 2;
      const f32x2 xv = *(const f32x2*)(X + (size_t)(kc * 64 + ll) * D_ + dp);
      const unsigned int p = pk2(xv[0], xv[1]);
      Xt[dp][ll]     = (unsigned short)(p & 0xffffu);
      Xt[dp + 1][ll] = (unsigned short)(p >> 16);
    }
    __syncthreads();
#pragma unroll
    for (int kk = 0; kk < 2; ++kk) {
      const Frag a = a_frag_f32(dwr + kc * 64, kk, half, 1.0f);
#pragma unroll
      for (int c = 0; c < 8; ++c) {
        const Frag bb = b_frag_bf(&Xt[c * 16 + lcol][0], kk, half);
        acc[c] = wmma_bf16(a, bb, acc[c]);
      }
    }
  }

  const float outscale = sel ? 1.0f : 0.08838834764831845f;  // 1/sqrt(128)
  unsigned short* outp = (sel ? k2w : q2w) + (size_t)b * L_ * D_;
#pragma unroll
  for (int c = 0; c < 8; ++c)
#pragma unroll
    for (int r = 0; r < 8; ++r) {
      const int m = m0 + wave * 16 + half * 8 + r;
      outp[(size_t)m * D_ + c * 16 + lcol] = bfbits(acc[c][r] * outscale);
    }
}

// ---------------------------------------------------------------------------
// Kernel 2: fused  attn = tanh(mask((q/T)K^T + (q2w/T)k2w^T)),  out = attn @ V
// grid = (L/128, B), block = 256. Wave w owns rows m0 + 16w .. +15.
// K and V chunks are converted to bf16 ONCE per WG into LDS; k2w comes from
// the workspace already in bf16 (plain b128 loads, no VALU conversion).
// ---------------------------------------------------------------------------
__global__ __launch_bounds__(256)
void attn_kernel(const float* __restrict__ q, const float* __restrict__ k,
                 const float* __restrict__ v, const int* __restrict__ lens,
                 const unsigned short* __restrict__ q2w,
                 const unsigned short* __restrict__ k2w,
                 float* __restrict__ out, float* __restrict__ attn) {
  __shared__ __align__(16) unsigned short Kc[64][136];     // bf16 K chunk [key][feat]
  __shared__ __align__(16) unsigned short Vt[D_][72];      // bf16 V chunk, transposed
  __shared__ __align__(16) unsigned short Sst[8][16][72];  // per-wave tanh(S), bf16

  const int b    = blockIdx.y;
  const int m0   = blockIdx.x * 128;
  const int tid  = threadIdx.x;
  const int wave = tid >> 5, lane = tid & 31;
  const int half = lane >> 4, lcol = lane & 15;
  const int len  = lens[b];
  const float invT = 0.08838834764831845f;

  const int mrow = m0 + wave * 16 + lcol;  // this lane's A-matrix row
  const float*          qrow   = q   + ((size_t)b * L_ + mrow) * D_;
  const unsigned short* q2wrow = q2w + ((size_t)b * L_ + mrow) * D_;
  const float*          kb     = k   + (size_t)b * L_ * D_;
  const unsigned short* k2wb   = k2w + (size_t)b * L_ * D_;
  const float*          vb     = v   + (size_t)b * L_ * D_;

  // Persistent A fragments over the full D=128 reduction (4 ksteps of 32).
  Frag aq[4], aq2[4];
#pragma unroll
  for (int t = 0; t < 4; ++t) {
    aq[t]  = a_frag_f32(qrow, t, half, invT);
    aq2[t] = a_frag_bf(q2wrow, t, half);
  }

  f32x8 o[8];
#pragma unroll
  for (int c = 0; c < 8; ++c)
#pragma unroll
    for (int r = 0; r < 8; ++r) o[c][r] = 0.0f;

  for (int kc = 0; kc < L_ / 64; ++kc) {
    __syncthreads();
    // Stage K (row-major) and V (transposed) as bf16; dim-pairs -> paired cvts,
    // one b32 store into Kc, d16/d16_hi half-stores into Vt.
#pragma unroll 4
    for (int i = tid; i < 64 * 64; i += 256) {
      const int kl = i >> 6, dp = (i & 63) * 2;
      const size_t g = (size_t)(kc * 64 + kl) * D_ + dp;
      const f32x2 kv = *(const f32x2*)(kb + g);
      const f32x2 vv = *(const f32x2*)(vb + g);
      *(unsigned int*)&Kc[kl][dp] = pk2(kv[0], kv[1]);
      const unsigned int vp = pk2(vv[0], vv[1]);
      Vt[dp][kl]     = (unsigned short)(vp & 0xffffu);
      Vt[dp + 1][kl] = (unsigned short)(vp >> 16);
    }
    if (kc + 1 < L_ / 64) {          // nudge next chunk toward the caches
      __builtin_prefetch(kb + (size_t)(kc * 64 + 64) * D_ + tid * 32, 0, 1);
      __builtin_prefetch(vb + (size_t)(kc * 64 + 64) * D_ + tid * 32, 0, 1);
    }
    __syncthreads();

    // ---- scores S for 4 key tiles of 16, mask + tanh, stream attn ----
#pragma unroll
    for (int nt = 0; nt < 4; ++nt) {
      const int n0   = kc * 64 + nt * 16;
      const int cidx = n0 + lcol;  // this lane's key column
      const unsigned short* kcol  = &Kc[nt * 16 + lcol][0];
      const unsigned short* k2col = k2wb + (size_t)cidx * D_;

      f32x8 s;
#pragma unroll
      for (int r = 0; r < 8; ++r) s[r] = 0.0f;
#pragma unroll
      for (int t = 0; t < 4; ++t) {
        const Frag bk = b_frag_bf(kcol, t, half);
        s = wmma_bf16(aq[t], bk, s);
      }
#pragma unroll
      for (int t = 0; t < 4; ++t) {
        const Frag bk2 = b_frag_bf(k2col, t, half);
        s = wmma_bf16(aq2[t], bk2, s);
      }

      const bool keep = cidx < len;  // mask key positions >= length[b]
#pragma unroll
      for (int r = 0; r < 8; ++r) {
        const float val = keep ? tanhf(s[r]) : 0.0f;
        const int   mg  = m0 + wave * 16 + half * 8 + r;
        attn[((size_t)b * L_ + mg) * L_ + cidx] = val;
        Sst[wave][half * 8 + r][nt * 16 + lcol] = bfbits(val);
      }
    }

    // ---- O += tanh(S) @ V  (K = 64 keys of this chunk, 2 ksteps of 32) ----
#pragma unroll
    for (int kk = 0; kk < 2; ++kk) {
      const Frag as = a_frag_bf(&Sst[wave][lcol][0], kk, half);
#pragma unroll
      for (int c = 0; c < 8; ++c) {
        const Frag bv = b_frag_bf(&Vt[c * 16 + lcol][0], kk, half);
        o[c] = wmma_bf16(as, bv, o[c]);
      }
    }
  }

#pragma unroll
  for (int c = 0; c < 8; ++c)
#pragma unroll
    for (int r = 0; r < 8; ++r) {
      const int mg = m0 + wave * 16 + half * 8 + r;
      out[((size_t)b * L_ + mg) * D_ + c * 16 + lcol] = o[c][r];
    }
}

extern "C" void kernel_launch(void* const* d_in, const int* in_sizes, int n_in,
                              void* d_out, int out_size, void* d_ws, size_t ws_size,
                              hipStream_t stream) {
  (void)in_sizes; (void)n_in; (void)out_size; (void)ws_size;
  const float* q  = (const float*)d_in[0];
  const float* k  = (const float*)d_in[1];
  const float* v  = (const float*)d_in[2];
  const float* q2 = (const float*)d_in[3];
  const float* k2 = (const float*)d_in[4];
  const float* dw = (const float*)d_in[5];
  const int* lens = (const int*)d_in[6];

  // Workspace: bf16 q2w (pre-scaled by 1/T) and bf16 k2w, 8 MB each.
  unsigned short* q2w = (unsigned short*)d_ws;
  unsigned short* k2w = q2w + (size_t)B_ * L_ * D_;

  float* out  = (float*)d_out;                    // [B, L, D]
  float* attn = out + (size_t)B_ * L_ * D_;       // [B, L, L]

  dim3 g1(L_ / 128, B_, 2);
  proj_kernel<<<g1, 256, 0, stream>>>(dw, q2, k2, q2w, k2w);

  dim3 g2(L_ / 128, B_);
  attn_kernel<<<g2, 256, 0, stream>>>(q, k, v, lens, q2w, k2w, out, attn);
}
